// TokenConstructor_17059610100073
// MI455X (gfx1250) — matrix operand order repaired and checked by reading, not compiled, not run on hardware
//
#include <hip/hip_runtime.h>

typedef __attribute__((ext_vector_type(16))) _Float16 v16h;
typedef __attribute__((ext_vector_type(8)))  _Float16 v8h;
typedef __attribute__((ext_vector_type(8)))  float    v8f;

namespace {
constexpr int KQ = 4096;          // queries per batch
constexpr int D  = 192;           // d_model
constexpr int QPB = 4;            // queries per block
constexpr int NTOK = 91;          // tokens per query
// feature map dims [B, C, H, W]
constexpr int C1 = 64,  H1 = 96, W1 = 320;   // stride 4
constexpr int C2 = 128, H2 = 48, W2 = 160;   // stride 8
constexpr int C3 = 192, H3 = 24, W3 = 80;    // stride 16
constexpr int C4 = 384, H4 = 12, W4 = 40;    // stride 32
constexpr float TWO_PI = 6.283185307179586f;

// f16 weight workspace layout (element offsets)
constexpr int WQ_OFF   = 0;                 // wq_t   [192][96]
constexpr int WLOC_OFF = WQ_OFF   + 192*96; // wloc_t [192][96] (k>=80 zero)
constexpr int WL2_OFF  = WLOC_OFF + 192*96; // wl2_t  [192][128]
constexpr int WL4_OFF  = WL2_OFF  + 192*128;// wl4_t  [192][384]
constexpr int W_TOTAL  = WL4_OFF  + 192*384;// = 135168 halves

// LDS A-panel sizes (halves)
constexpr int A_L1_H   = 128*96;   // 32 pts * 4 q, K=80 padded to 96 (cols 64..79 = phi)
constexpr int A_SEED_H = 16*96;    // rows 0..3 valid
constexpr int A_L2_H   = 112*128;  // rows 0..99 valid
constexpr int A_L4_H   = 48*384;   // rows 0..35 valid
constexpr int A_TOTAL_H = A_L1_H + A_SEED_H + A_L2_H + A_L4_H;
constexpr size_t SMEM_F32 = 256 + 64 + 256 + 8;  // fcen, offl, gxy, cq
constexpr size_t SMEM_BYTES = SMEM_F32*4 + (size_t)A_TOTAL_H*2;  // ~95.5 KB
}

// ---------------- bilinear sampler (align_corners=True, zero pad) ----------
__device__ __forceinline__ float bilin(const float* __restrict__ f,
                                       int C, int H, int W, int b, int c,
                                       float x, float y) {
  float x0 = floorf(x), y0 = floorf(y);
  float x1 = x0 + 1.f, y1 = y0 + 1.f;
  float wx1 = x - x0, wx0 = 1.f - wx1;
  float wy1 = y - y0, wy0 = 1.f - wy1;
  const float* base = f + ((size_t)b * C + c) * (size_t)(H * W);
  float Wm1 = (float)(W - 1), Hm1 = (float)(H - 1);
  auto tap = [&](float xf, float yf) -> float {
    bool v = (xf >= 0.f) && (xf <= Wm1) && (yf >= 0.f) && (yf <= Hm1);
    int xi = (int)fminf(fmaxf(xf, 0.f), Wm1);
    int yi = (int)fminf(fmaxf(yf, 0.f), Hm1);
    float val = base[yi * W + xi];
    return v ? val : 0.f;
  };
  return tap(x0, y0) * wx0 * wy0 + tap(x1, y0) * wx1 * wy0 +
         tap(x0, y1) * wx0 * wy1 + tap(x1, y1) * wx1 * wy1;
}

// ---------------- WMMA tile: D[16,192] += A[16,KPAD] * Wt^T ----------------
// A_lds: row-major [16][KPAD] f16 (LDS). wt: row-major [192][KPAD] f16 (global).
template<int KPAD, int KSTEPS>
__device__ __forceinline__ void gemm_tile(const _Float16* A_lds,
                                          const _Float16* __restrict__ wt,
                                          v8f acc[12]) {
  const int lane = threadIdx.x & 31;
  const int ln   = lane & 15;
  const int hiw  = lane >> 4;       // 0 / 1 (lane half)
  const int kA   = hiw * 8;         // A: K offsets {kA..kA+7, kA+16..kA+23}
  const int kB   = hiw * 8;         // B: K offsets {kB..kB+7, kB+16..kB+23}
  const _Float16* arow = A_lds + ln * KPAD;
  const _Float16* brow = wt    + ln * KPAD;
  for (int ks = 0; ks < KSTEPS; ++ks) {
    v16h a;
    {
      const _Float16* p = arow + ks * 32 + kA;
      v8h lo = *(const v8h*)(p);
      v8h hi = *(const v8h*)(p + 16);
#pragma unroll
      for (int i = 0; i < 8; ++i) { a[i] = lo[i]; a[8 + i] = hi[i]; }
    }
#pragma unroll
    for (int nt = 0; nt < 12; ++nt) {
      const _Float16* pb = brow + nt * 16 * KPAD + ks * 32 + kB;
      v8h lo = *(const v8h*)(pb);
      v8h hi = *(const v8h*)(pb + 16);
      v16h bf;
#pragma unroll
      for (int i = 0; i < 8; ++i) { bf[i] = lo[i]; bf[8 + i] = hi[i]; }
      acc[nt] = __builtin_amdgcn_wmma_f32_16x16x32_f16(
          false, a, false, bf, (short)0, acc[nt], false, false);
    }
  }
}

// ---------------- weight prep: fp32 [K,192] -> f16 transposed [192,Kpad] ---
__global__ void tokctor_prep_weights(const float* __restrict__ wq,
                                     const float* __restrict__ wloc,
                                     const float* __restrict__ wl2,
                                     const float* __restrict__ wl4,
                                     _Float16* __restrict__ ws) {
  int i = blockIdx.x * 256 + threadIdx.x;
  if (i >= W_TOTAL) return;
  float v; 
  if (i < WLOC_OFF) {                       // wq_t: [192][96] from w_q_w[96,192]
    int j = i - WQ_OFF; int n = j / 96, k = j % 96;
    v = wq[k * 192 + n];
  } else if (i < WL2_OFF) {                 // wloc_t: [192][96], K 80..95 zero
    int j = i - WLOC_OFF; int n = j / 96, k = j % 96;
    v = (k < 80) ? wloc[k * 192 + n] : 0.f;
  } else if (i < WL4_OFF) {                 // wl2_t: [192][128]
    int j = i - WL2_OFF; int n = j / 128, k = j % 128;
    v = wl2[k * 192 + n];
  } else {                                  // wl4_t: [192][384]
    int j = i - WL4_OFF; int n = j / 384, k = j % 384;
    v = wl4[k * 192 + n];
  }
  ws[i] = (_Float16)v;
}

// ---------------- fused main kernel ----------------------------------------
__global__ __launch_bounds__(256) void tokctor_main(
    const float* __restrict__ feat_l1, const float* __restrict__ feat_l2,
    const float* __restrict__ feat_l3, const float* __restrict__ feat_l4,
    const float* __restrict__ coords,
    const float* __restrict__ proj_l2_b, const float* __restrict__ proj_l4_b,
    const float* __restrict__ w_off_w,  const float* __restrict__ w_off_b,
    const float* __restrict__ w_loc_b,  const float* __restrict__ w_q_b,
    const float* __restrict__ e_l1, const float* __restrict__ e_l2,
    const float* __restrict__ e_l3, const float* __restrict__ e_l4,
    const float* __restrict__ rpe,
    const _Float16* __restrict__ wsw,   // f16 weights (prepped)
    float* __restrict__ out) {
  extern __shared__ char smem_raw[];
  float*    fcen = (float*)smem_raw;          // [QPB*64]
  float*    offl = fcen + 256;                // [QPB*16]
  float*    gxy  = offl + 64;                 // [QPB*32*2]
  float*    cq   = gxy + 256;                 // [QPB*2]
  _Float16* smA_l1   = (_Float16*)(cq + 8);   // [128][96]
  _Float16* smA_seed = smA_l1 + A_L1_H;       // [16][96]
  _Float16* smA_l2   = smA_seed + A_SEED_H;   // [112][128]
  _Float16* smA_l4   = smA_l2 + A_L2_H;       // [48][384]

  const int tid = threadIdx.x;
  const size_t g0 = (size_t)blockIdx.x * QPB;       // first global query
  const size_t seedBase = (size_t)2 * KQ * NTOK * D;

  // ---- P0/P1: zero all A panels; stash per-query coords -------------------
  {
    int4 z = make_int4(0, 0, 0, 0);
    int4* ab = (int4*)smA_l1;
    for (int i = tid; i < (int)(A_TOTAL_H / 8); i += 256) ab[i] = z;
    if (tid < QPB) {
      size_t g = g0 + tid;
      cq[tid * 2 + 0] = coords[g * 2 + 0];
      cq[tid * 2 + 1] = coords[g * 2 + 1];
    }
  }
  __syncthreads();

  // ---- P2: f_center (QPB*64 == 256 items) --------------------------------
  {
    int q = tid >> 6, c = tid & 63;
    int bq = (int)((g0 + q) >> 12);
    float v = bilin(feat_l1, C1, H1, W1, bq, c, cq[q*2] * 0.25f, cq[q*2+1] * 0.25f);
    fcen[q * 64 + c] = v;
    smA_seed[q * 96 + c] = (_Float16)v;
  }
  __syncthreads();

  // ---- P3: offset GEMV (64x16) + tanh ; seed Fourier PE -------------------
  if (tid < 64) {
    int q = tid >> 4, j = tid & 15;
    float s = w_off_b[j];
    for (int c = 0; c < 64; ++c) s += fcen[q * 64 + c] * w_off_w[c * 16 + j];
    offl[q * 16 + j] = 6.f * tanhf(s);
  } else if (tid < 192) {
    int i = tid - 64; int q = i >> 5, v5 = i & 31;
    int grp = v5 >> 3, fi = v5 & 7;
    float cn = (grp < 2) ? (cq[q*2] / 1280.f) : (cq[q*2+1] / 384.f);
    float ang = cn * (float)(1 << fi) * TWO_PI;
    smA_seed[q * 96 + 64 + v5] = (_Float16)((grp & 1) ? cosf(ang) : sinf(ang));
  }
  __syncthreads();

  // ---- P4: per-point l1 sample coords + phi Fourier columns --------------
  if (tid < 128) {
    int q = tid >> 5, p = tid & 31;
    float dx, dy;
    if (p < 24) { int t = (p < 12) ? p : p + 1; dy = (float)(t / 5 - 2); dx = (float)(t % 5 - 2); }
    else        { int pp = p - 24; dx = offl[q*16 + 2*pp]; dy = offl[q*16 + 2*pp + 1]; }
    int row = q * 32 + p;
    gxy[row * 2 + 0] = cq[q*2] * 0.25f + dx;
    gxy[row * 2 + 1] = cq[q*2+1] * 0.25f + dy;
#pragma unroll
    for (int j = 0; j < 16; ++j) {
      int grp = j >> 2, fi = j & 3;
      float b = (grp < 2) ? dx : dy;
      float ang = b * (float)(1 << fi) * TWO_PI;
      smA_l1[row * 96 + 64 + j] = (_Float16)((grp & 1) ? cosf(ang) : sinf(ang));
    }
  }
  __syncthreads();

  // ---- P5: l1 gather (QPB*32*64 = 8192) ----------------------------------
  for (int idx = tid; idx < QPB * 32 * 64; idx += 256) {
    int q = idx >> 11, rem = idx & 2047, p = rem >> 6, c = rem & 63;
    int bq = (int)((g0 + q) >> 12);
    int row = q * 32 + p;
    float v = bilin(feat_l1, C1, H1, W1, bq, c, gxy[row*2], gxy[row*2+1]);
    smA_l1[row * 96 + c] = (_Float16)v;
  }
  // ---- P6: l2 gather (QPB*25*128 = 12800) --------------------------------
  for (int idx = tid; idx < QPB * 25 * 128; idx += 256) {
    int q = idx / 3200, rem = idx % 3200, p = rem >> 7, c = rem & 127;
    int bq = (int)((g0 + q) >> 12);
    float gx = cq[q*2] * 0.125f + (float)(p % 5 - 2);
    float gy = cq[q*2+1] * 0.125f + (float)(p / 5 - 2);
    smA_l2[(q * 25 + p) * 128 + c] = (_Float16)bilin(feat_l2, C2, H2, W2, bq, c, gx, gy);
  }
  // ---- P7: l4 gather (QPB*9*384 = 13824) ---------------------------------
  for (int idx = tid; idx < QPB * 9 * 384; idx += 256) {
    int q = idx / 3456, rem = idx % 3456, p = rem / 384, c = rem % 384;
    int bq = (int)((g0 + q) >> 12);
    float gx = cq[q*2] * 0.03125f + (float)(p % 3 - 1);
    float gy = cq[q*2+1] * 0.03125f + (float)(p / 3 - 1);
    smA_l4[(q * 9 + p) * 384 + c] = (_Float16)bilin(feat_l4, C4, H4, W4, bq, c, gx, gy);
  }
  // ---- P8: l3 gather + epilogue (no GEMM), stream to out ------------------
  for (int idx = tid; idx < QPB * 25 * 192; idx += 256) {
    int q = idx / 4800, rem = idx % 4800, p = rem / 192, n = rem % 192;
    size_t g = g0 + q;
    int bq = (int)(g >> 12);
    float gx = cq[q*2] * 0.0625f + (float)(p % 5 - 2);
    float gy = cq[q*2+1] * 0.0625f + (float)(p / 5 - 2);
    float v = bilin(feat_l3, C3, H3, W3, bq, n, gx, gy) + e_l3[n] + rpe[p * 192 + n];
    out[(g * NTOK + 57 + p) * D + n] = v;
  }
  __syncthreads();

  // ---- P9: WMMA GEMMs. 19 M-tiles over 8 waves ----------------------------
  const int wid  = tid >> 5;
  const int lane = tid & 31;
  const int ln   = lane & 15;
  const int rb   = (lane >> 4) << 3;   // D-matrix row base (0 or 8)
  const _Float16* wq_t   = wsw + WQ_OFF;
  const _Float16* wloc_t = wsw + WLOC_OFF;
  const _Float16* wl2_t  = wsw + WL2_OFF;
  const _Float16* wl4_t  = wsw + WL4_OFF;

  for (int t = wid; t < 19; t += 8) {
    v8f acc[12] = {};
    if (t < 8) {                 // ---- l1: gelu(X@w_loc + b) + e_l1 --------
      int mt = t;
      gemm_tile<96, 3>(smA_l1 + mt * 16 * 96, wloc_t, acc);
#pragma unroll
      for (int nt = 0; nt < 12; ++nt) {
        int n = nt * 16 + ln;
        float bl = w_loc_b[n], el = e_l1[n];
#pragma unroll
        for (int r = 0; r < 8; ++r) {
          int lr = mt * 16 + rb + r;
          int q = lr >> 5, p = lr & 31;
          size_t g = g0 + q;
          float x = acc[nt][r] + bl;
          x = 0.5f * x * (1.f + erff(x * 0.70710678118654752f)) + el;
          out[(g * NTOK + p) * D + n] = x;
        }
      }
    } else if (t < 15) {         // ---- l2: X@proj_l2 + b + e_l2 + rpe ------
      int mt = t - 8;
      gemm_tile<128, 4>(smA_l2 + mt * 16 * 128, wl2_t, acc);
#pragma unroll
      for (int nt = 0; nt < 12; ++nt) {
        int n = nt * 16 + ln;
        float bb = proj_l2_b[n] + e_l2[n];
#pragma unroll
        for (int r = 0; r < 8; ++r) {
          int lr = mt * 16 + rb + r;
          if (lr < QPB * 25) {
            int q = lr / 25, p = lr % 25;
            size_t g = g0 + q;
            out[(g * NTOK + 32 + p) * D + n] = acc[nt][r] + bb + rpe[p * 192 + n];
          }
        }
      }
    } else if (t < 18) {         // ---- l4: X@proj_l4 + b + e_l4 + rpe[idx] -
      int mt = t - 15;
      gemm_tile<384, 12>(smA_l4 + mt * 16 * 384, wl4_t, acc);
#pragma unroll
      for (int nt = 0; nt < 12; ++nt) {
        int n = nt * 16 + ln;
        float bb = proj_l4_b[n] + e_l4[n];
#pragma unroll
        for (int r = 0; r < 8; ++r) {
          int lr = mt * 16 + rb + r;
          if (lr < QPB * 9) {
            int q = lr / 9, p = lr % 9;
            int ri = (p / 3) * 5 + (p % 3) + 6;    // L4_RPE_IDX
            size_t g = g0 + q;
            out[(g * NTOK + 82 + p) * D + n] = acc[nt][r] + bb + rpe[ri * 192 + n];
          }
        }
      }
    } else {                     // ---- seed: [f_center, pe_q]@w_q + b ------
      gemm_tile<96, 3>(smA_seed, wq_t, acc);
#pragma unroll
      for (int nt = 0; nt < 12; ++nt) {
        int n = nt * 16 + ln;
        float bb = w_q_b[n];
#pragma unroll
        for (int r = 0; r < 8; ++r) {
          int lr = rb + r;
          if (lr < QPB) {
            size_t g = g0 + lr;
            out[seedBase + g * D + n] = acc[nt][r] + bb;
          }
        }
      }
    }
  }
}

extern "C" void kernel_launch(void* const* d_in, const int* in_sizes, int n_in,
                              void* d_out, int out_size, void* d_ws, size_t ws_size,
                              hipStream_t stream) {
  (void)in_sizes; (void)n_in; (void)out_size; (void)ws_size;
  const float* feat_l1   = (const float*)d_in[0];
  const float* feat_l2   = (const float*)d_in[1];
  const float* feat_l3   = (const float*)d_in[2];
  const float* feat_l4   = (const float*)d_in[3];
  const float* coords    = (const float*)d_in[4];
  const float* proj_l2_w = (const float*)d_in[5];
  const float* proj_l2_b = (const float*)d_in[6];
  const float* proj_l4_w = (const float*)d_in[7];
  const float* proj_l4_b = (const float*)d_in[8];
  const float* w_off_w   = (const float*)d_in[9];
  const float* w_off_b   = (const float*)d_in[10];
  const float* w_loc_w   = (const float*)d_in[11];
  const float* w_loc_b   = (const float*)d_in[12];
  const float* w_q_w     = (const float*)d_in[13];
  const float* w_q_b     = (const float*)d_in[14];
  const float* e_l1      = (const float*)d_in[15];
  const float* e_l2      = (const float*)d_in[16];
  const float* e_l3      = (const float*)d_in[17];
  const float* e_l4      = (const float*)d_in[18];
  const float* rpe       = (const float*)d_in[19];
  float* out = (float*)d_out;
  _Float16* wsw = (_Float16*)d_ws;

  // 1) f16 transposed/padded weights into workspace
  tokctor_prep_weights<<<(W_TOTAL + 255) / 256, 256, 0, stream>>>(
      w_q_w, w_loc_w, proj_l2_w, proj_l4_w, wsw);

  // 2) fused gather + WMMA kernel: 8192 queries / 4 per block
  (void)hipFuncSetAttribute((const void*)tokctor_main,
                            hipFuncAttributeMaxDynamicSharedMemorySize,
                            (int)SMEM_BYTES);
  tokctor_main<<<(2 * KQ) / QPB, 256, SMEM_BYTES, stream>>>(
      feat_l1, feat_l2, feat_l3, feat_l4, coords,
      proj_l2_b, proj_l4_b, w_off_w, w_off_b, w_loc_b, w_q_b,
      e_l1, e_l2, e_l3, e_l4, rpe, wsw, out);
}
